// MultiHeadAttention_21251498181338
// MI455X (gfx1250) — matrix operand order repaired and checked by reading, not compiled, not run on hardware
//
#include <hip/hip_runtime.h>
#include <hip/hip_bf16.h>

typedef __bf16 bf16;
typedef __attribute__((ext_vector_type(16))) __bf16 v16bf;
typedef __attribute__((ext_vector_type(8)))  __bf16 v8bf;
typedef __attribute__((ext_vector_type(8)))  float  v8f;

#define D_MODEL  1024
#define NUM_HEADS 16
#define DEPTH      64
#define SEQ_L    2048
#define BATCH       2
#define ROWS_M   (BATCH * SEQ_L)   // 4096

// ---------------------------------------------------------------------------
// WMMA fragment helpers (wave32, 16x16x32 bf16 -> f32)
// A (16x32): lane m = lane&15; half = lane>>4 selects K subsets {0..7,16..23}
//            (half=0) / {8..15,24..31} (half=1); 16 bf16 per lane.
// B (32x16): lane n = lane&15; same K subsets as A ({0..7,16..23} / {8..15,24..31})
//            along K  (requires K-major source, hence transposed weights).
// C (16x16): elem v -> row m = v + (lane>>4)*8, col n = lane&15.
// ---------------------------------------------------------------------------
static __device__ __forceinline__ v16bf load_a_frag(const bf16* base, int ld) {
  int lane = threadIdx.x & 31;
  int m = lane & 15;
  int half = lane >> 4;
  const bf16* p = base + (size_t)m * ld + half * 8;
  v8bf lo = *(const v8bf*)(p);
  v8bf hi = *(const v8bf*)(p + 16);
  v16bf r;
#pragma unroll
  for (int i = 0; i < 8; ++i) { r[i] = lo[i]; r[i + 8] = hi[i]; }
  return r;
}

static __device__ __forceinline__ v16bf load_b_frag(const bf16* base, int ld) {
  int lane = threadIdx.x & 31;
  int n = lane & 15;
  int kb = (lane >> 4) * 8;
  const bf16* p = base + (size_t)n * ld + kb;
  v8bf lo = *(const v8bf*)(p);
  v8bf hi = *(const v8bf*)(p + 16);
  v16bf r;
#pragma unroll
  for (int i = 0; i < 8; ++i) { r[i] = lo[i]; r[i + 8] = hi[i]; }
  return r;
}

static __device__ __forceinline__ v8f wmma_bf16(v16bf a, v16bf b, v8f c) {
  return __builtin_amdgcn_wmma_f32_16x16x32_bf16(false, a, false, b,
                                                 (short)0, c, false, false);
}

// ---------------------------------------------------------------------------
// fp32 -> bf16 casts / layout shuffles
// ---------------------------------------------------------------------------
__global__ void cast_bf16_kernel(const float* __restrict__ src,
                                 bf16* __restrict__ dst, int n) {
  int i = blockIdx.x * blockDim.x + threadIdx.x;
  if (i < n) dst[i] = (bf16)src[i];
}

// Wt[n*1024 + k] = W[k*1024 + n]   (so B-fragments read contiguous K)
__global__ void transpose_cast_kernel(const float* __restrict__ W,
                                      bf16* __restrict__ Wt) {
  int i = blockIdx.x * blockDim.x + threadIdx.x;
  if (i < D_MODEL * D_MODEL) {
    int n = i >> 10;
    int k = i & 1023;
    Wt[i] = (bf16)W[(size_t)k * D_MODEL + n];
  }
}

// Erev[h][r][d] = E[L-1-r][h*64+d]   (head-split, row-reversed)
__global__ void erev_cast_kernel(const float* __restrict__ E,
                                 bf16* __restrict__ Erev) {
  int i = blockIdx.x * blockDim.x + threadIdx.x;
  if (i < NUM_HEADS * SEQ_L * DEPTH) {
    int d = i & 63;
    int r = (i >> 6) & (SEQ_L - 1);
    int h = i >> 17;
    Erev[i] = (bf16)E[(size_t)(SEQ_L - 1 - r) * D_MODEL + h * DEPTH + d];
  }
}

// ---------------------------------------------------------------------------
// GEMM: out = A[M,K](bf16) x Wt[N,K](bf16, K-major) + bias
//   mode 0: bf16 out, head-split   [B,H,L,64]
//   mode 1: bf16 out, head-split V-transposed [B,H,64,L]
//   mode 2: fp32 out, row-major    [M,1024]
// WG = 128 threads (4 waves); tile 128x64; wave w owns rows [tileM+32w, +32):
// two A-row fragments share each B-fragment -> 8 WMMA per (2A + 4B) loads.
// ---------------------------------------------------------------------------
__global__ __launch_bounds__(128) void gemm_kernel(const bf16* __restrict__ A,
                                                   const bf16* __restrict__ Wt,
                                                   const float* __restrict__ bias,
                                                   bf16* __restrict__ outb,
                                                   float* __restrict__ outf,
                                                   int K, int mode) {
  int tileN = blockIdx.x * 64;
  int tileM = blockIdx.y * 128;
  int wave = threadIdx.x >> 5;
  int lane = threadIdx.x & 31;
  int half = lane >> 4;
  int m0 = tileM + wave * 32;

  v8f acc[2][4] = {};
  const bf16* Arow0 = A + (size_t)m0 * K;
  const bf16* Arow1 = Arow0 + (size_t)16 * K;
  for (int k0 = 0; k0 < K; k0 += 32) {
    __builtin_prefetch(Arow0 + k0 + 64, 0, 1);   // global_prefetch_b8
    __builtin_prefetch(Arow1 + k0 + 64, 0, 1);
    v16bf a0 = load_a_frag(Arow0 + k0, K);
    v16bf a1 = load_a_frag(Arow1 + k0, K);
#pragma unroll
    for (int j = 0; j < 4; ++j) {
      v16bf b = load_b_frag(Wt + (size_t)(tileN + j * 16) * K + k0, K);
      acc[0][j] = wmma_bf16(a0, b, acc[0][j]);
      acc[1][j] = wmma_bf16(a1, b, acc[1][j]);
    }
  }

#pragma unroll
  for (int r = 0; r < 2; ++r) {
#pragma unroll
    for (int j = 0; j < 4; ++j) {
#pragma unroll
      for (int v = 0; v < 8; ++v) {
        int gm = m0 + r * 16 + v + half * 8;
        int gn = tileN + j * 16 + (lane & 15);
        float val = acc[r][j][v] + bias[gn];
        if (mode == 2) {
          outf[(size_t)gm * D_MODEL + gn] = val;
        } else {
          int b = gm >> 11, l = gm & (SEQ_L - 1);
          int h = gn >> 6,  d = gn & 63;
          size_t idx = (mode == 0)
            ? (((size_t)b * NUM_HEADS + h) * SEQ_L + l) * DEPTH + d
            : (((size_t)b * NUM_HEADS + h) * DEPTH + d) * SEQ_L + l;
          outb[idx] = (bf16)val;
        }
      }
    }
  }
}

// ---------------------------------------------------------------------------
// Attention: one 16-row q-tile per WG. LDS holds full fp32 logits row-block
// (16 x 2048 = 128 KB of the 320 KB WGP pool), per-wave R-tile scratch, and
// softmax reduction slots.
//   logits[q,k] = (Q.K^T + R[q, q-k]) / 8,  R = Q.Erev^T  (skew == diag gather)
// ---------------------------------------------------------------------------
__global__ __launch_bounds__(128) void attn_kernel(const bf16* __restrict__ Qh,
                                                   const bf16* __restrict__ Kh,
                                                   const bf16* __restrict__ Vht,
                                                   const bf16* __restrict__ Erev,
                                                   bf16* __restrict__ ctx) {
  extern __shared__ float smem[];
  float* P      = smem;                 // 16 * 2048
  float* rsc    = P + 16 * SEQ_L;       // 4 waves * 2 tiles * 256
  float* red    = rsc + 4 * 512;        // 16 * 8
  float* rowsum = red + 128;            // 16

  int bh = blockIdx.x >> 7;             // (b*16 + h)
  int qt = (blockIdx.x & 127) << 4;     // q-tile base
  int h  = bh & 15;
  int wave = threadIdx.x >> 5;
  int lane = threadIdx.x & 31;
  int n16  = lane & 15;
  int half = lane >> 4;

  const bf16* Qb = Qh   + (size_t)bh * SEQ_L * DEPTH;
  const bf16* Kb = Kh   + (size_t)bh * SEQ_L * DEPTH;
  const bf16* Vb = Vht  + (size_t)bh * DEPTH * SEQ_L;
  const bf16* Eb = Erev + (size_t)h  * SEQ_L * DEPTH;

  // Q A-fragments (shared by all k-subtiles of this wave)
  v16bf aq0 = load_a_frag(Qb + (size_t)qt * DEPTH, DEPTH);
  v16bf aq1 = load_a_frag(Qb + (size_t)qt * DEPTH + 32, DEPTH);

  float* myrs = rsc + wave * 512;

  // ---- pass 1: logits -------------------------------------------------
  for (int ks = wave * 16; ks <= qt; ks += 64) {
    int d = qt - ks;                       // multiple of 16, >= 0
    v8f cs = {};
    cs = wmma_bf16(aq0, load_b_frag(Kb + (size_t)ks * DEPTH,      DEPTH), cs);
    cs = wmma_bf16(aq1, load_b_frag(Kb + (size_t)ks * DEPTH + 32, DEPTH), cs);
    v8f cd = {};                           // R cols [d, d+16)
    cd = wmma_bf16(aq0, load_b_frag(Eb + (size_t)d * DEPTH,      DEPTH), cd);
    cd = wmma_bf16(aq1, load_b_frag(Eb + (size_t)d * DEPTH + 32, DEPTH), cd);
    v8f cdm = {};                          // R cols [d-16, d)
    if (d >= 16) {
      cdm = wmma_bf16(aq0, load_b_frag(Eb + (size_t)(d - 16) * DEPTH,      DEPTH), cdm);
      cdm = wmma_bf16(aq1, load_b_frag(Eb + (size_t)(d - 16) * DEPTH + 32, DEPTH), cdm);
    }
#pragma unroll
    for (int v = 0; v < 8; ++v) {          // stage R tiles for diagonal gather
      int m = v + half * 8;
      myrs[m * 16 + n16]       = cd[v];
      myrs[256 + m * 16 + n16] = cdm[v];
    }
    asm volatile("s_wait_dscnt 0" ::: "memory");   // intra-wave LDS RAW
#pragma unroll
    for (int v = 0; v < 8; ++v) {
      int m = v + half * 8;
      int kcol = ks + n16;
      int qrow = qt + m;
      float val;
      if (kcol > qrow) {
        val = -1.0e30f;                    // causal mask
      } else {
        int diff = m - n16;                // r = d + diff
        float rv = (diff >= 0) ? myrs[m * 16 + diff]
                               : myrs[256 + m * 16 + 16 + diff];
        val = (cs[v] + rv) * 0.125f;       // 1/sqrt(64)
      }
      P[m * SEQ_L + kcol] = val;
    }
  }
  __syncthreads();

  // ---- pass 2: softmax over LDS row-block -----------------------------
  int k_end = qt + 16;
  int k_pad = (k_end + 31) & ~31;
  int row = threadIdx.x >> 3;              // 16 rows x 8 threads
  int c8  = threadIdx.x & 7;
  for (int k = k_end + c8; k < k_pad; k += 8) P[row * SEQ_L + k] = 0.0f;
  float mx = -1.0e30f;
  for (int k = c8; k < k_end; k += 8) mx = fmaxf(mx, P[row * SEQ_L + k]);
  red[row * 8 + c8] = mx;
  __syncthreads();
  float rmx = red[row * 8];
#pragma unroll
  for (int i = 1; i < 8; ++i) rmx = fmaxf(rmx, red[row * 8 + i]);
  float s = 0.0f;
  for (int k = c8; k < k_end; k += 8) {
    float e = __expf(P[row * SEQ_L + k] - rmx);
    P[row * SEQ_L + k] = e;
    s += e;
  }
  __syncthreads();
  red[row * 8 + c8] = s;
  __syncthreads();
  if (c8 == 0) {
    float rs = 0.0f;
#pragma unroll
    for (int i = 0; i < 8; ++i) rs += red[row * 8 + i];
    rowsum[row] = rs;
  }
  __syncthreads();

  // ---- pass 3: O = P x V  (wave w owns depth cols [16w, 16w+16)) ------
  int n0 = wave * 16;
  v8f acc = {};
  for (int kk = 0; kk < k_end; kk += 32) {
    v16bf ap;
    const float* prow = P + (size_t)n16 * SEQ_L + kk + half * 8;
#pragma unroll
    for (int i = 0; i < 8; ++i) {
      ap[i]     = (bf16)prow[i];
      ap[i + 8] = (bf16)prow[16 + i];
    }
    v16bf bv = load_b_frag(Vb + (size_t)n0 * SEQ_L + kk, SEQ_L);
    acc = wmma_bf16(ap, bv, acc);
  }
  int b = bh >> 4;
#pragma unroll
  for (int v = 0; v < 8; ++v) {
    int m = v + half * 8;
    int dcol = n0 + n16;
    float o = acc[v] / rowsum[m];
    ctx[((size_t)b * SEQ_L + qt + m) * D_MODEL + h * DEPTH + dcol] = (bf16)o;
  }
}

// ---------------------------------------------------------------------------
// Launcher
// ---------------------------------------------------------------------------
extern "C" void kernel_launch(void* const* d_in, const int* in_sizes, int n_in,
                              void* d_out, int out_size, void* d_ws, size_t ws_size,
                              hipStream_t stream) {
  const float* q  = (const float*)d_in[0];
  const float* k  = (const float*)d_in[1];
  const float* v  = (const float*)d_in[2];
  // d_in[3] = mask (causal applied analytically)
  const float* Wq = (const float*)d_in[4];
  const float* bq = (const float*)d_in[5];
  const float* Wk = (const float*)d_in[6];
  const float* bk = (const float*)d_in[7];
  const float* Wv = (const float*)d_in[8];
  const float* bv = (const float*)d_in[9];
  const float* Wo = (const float*)d_in[10];
  const float* bo = (const float*)d_in[11];
  const float* E  = (const float*)d_in[12];

  bf16* ws = (bf16*)d_ws;
  const size_t nAct = (size_t)ROWS_M * D_MODEL;     // 4 Mi elems
  const size_t nW   = (size_t)D_MODEL * D_MODEL;    // 1 Mi
  const size_t nE   = (size_t)NUM_HEADS * SEQ_L * DEPTH; // 2 Mi
  bf16* qb  = ws;
  bf16* kb  = qb  + nAct;
  bf16* vb  = kb  + nAct;
  bf16* WqT = vb  + nAct;
  bf16* WkT = WqT + nW;
  bf16* WvT = WkT + nW;
  bf16* WoT = WvT + nW;
  bf16* Erv = WoT + nW;
  bf16* Qh  = Erv + nE;
  bf16* Kh  = Qh  + nAct;
  bf16* Vt  = Kh  + nAct;
  bf16* Ctx = Vt  + nAct;

  const int T = 256;
  cast_bf16_kernel<<<(int)((nAct + T - 1) / T), T, 0, stream>>>(q, qb, (int)nAct);
  cast_bf16_kernel<<<(int)((nAct + T - 1) / T), T, 0, stream>>>(k, kb, (int)nAct);
  cast_bf16_kernel<<<(int)((nAct + T - 1) / T), T, 0, stream>>>(v, vb, (int)nAct);
  transpose_cast_kernel<<<(int)((nW + T - 1) / T), T, 0, stream>>>(Wq, WqT);
  transpose_cast_kernel<<<(int)((nW + T - 1) / T), T, 0, stream>>>(Wk, WkT);
  transpose_cast_kernel<<<(int)((nW + T - 1) / T), T, 0, stream>>>(Wv, WvT);
  transpose_cast_kernel<<<(int)((nW + T - 1) / T), T, 0, stream>>>(Wo, WoT);
  erev_cast_kernel<<<(int)((nE + T - 1) / T), T, 0, stream>>>(E, Erv);

  dim3 gg(D_MODEL / 64, ROWS_M / 128);   // (16, 32)
  gemm_kernel<<<gg, 128, 0, stream>>>(qb, WqT, bq, Qh, nullptr, D_MODEL, 0);
  gemm_kernel<<<gg, 128, 0, stream>>>(kb, WkT, bk, Kh, nullptr, D_MODEL, 0);
  gemm_kernel<<<gg, 128, 0, stream>>>(vb, WvT, bv, Vt, nullptr, D_MODEL, 1);

  size_t smem = (size_t)(16 * SEQ_L + 4 * 512 + 128 + 16) * sizeof(float);
  (void)hipFuncSetAttribute((const void*)attn_kernel,
                            hipFuncAttributeMaxDynamicSharedMemorySize, (int)smem);
  attn_kernel<<<BATCH * NUM_HEADS * (SEQ_L / 16), 128, smem, stream>>>(
      Qh, Kh, Vt, Erv, Ctx);

  gemm_kernel<<<gg, 128, 0, stream>>>(Ctx, WoT, bo, nullptr, (float*)d_out,
                                      D_MODEL, 2);
}
